// GCN_85349590106533
// MI455X (gfx1250) — matrix operand order repaired and checked by reading, not compiled, not run on hardware
//
#include <hip/hip_runtime.h>
#include <hip/hip_bf16.h>

#define N_NODES   50000
#define N_EDGES   800000
#define N_GRAPHS  256
#define D         64
#define HID       128
#define OUT_DIM   138

typedef __attribute__((ext_vector_type(16))) _Float16 v16h;
typedef __attribute__((ext_vector_type(8)))  _Float16 v8h;
__device__ __forceinline__ v16h ldB(const _Float16* p){
  const v8h lo = *(const v8h*)p, hi = *(const v8h*)(p + 16);
  return __builtin_shufflevector(lo, hi, 0,1,2,3,4,5,6,7,8,9,10,11,12,13,14,15);
}
typedef __attribute__((ext_vector_type(8)))  float    v8f;

__device__ __forceinline__ float wave_sum(float v){
#pragma unroll
  for (int m = 16; m > 0; m >>= 1) v += __shfl_xor(v, m, 32);
  return v;
}
__device__ __forceinline__ float wave_max(float v){
#pragma unroll
  for (int m = 16; m > 0; m >>= 1) v = fmaxf(v, __shfl_xor(v, m, 32));
  return v;
}
__device__ __forceinline__ float dot4(float4 a, float4 b, float acc){
  acc = fmaf(a.x, b.x, acc); acc = fmaf(a.y, b.y, acc);
  acc = fmaf(a.z, b.z, acc); acc = fmaf(a.w, b.w, acc);
  return acc;
}
// pack two float4 (8 consecutive K values) into fragment halves [off..off+7]
__device__ __forceinline__ void pack8(v16h& dst, int off, float4 p, float4 q){
  dst[off+0] = (_Float16)p.x; dst[off+1] = (_Float16)p.y;
  dst[off+2] = (_Float16)p.z; dst[off+3] = (_Float16)p.w;
  dst[off+4] = (_Float16)q.x; dst[off+5] = (_Float16)q.y;
  dst[off+6] = (_Float16)q.z; dst[off+7] = (_Float16)q.w;
}
// pack with fused y = relu(v*s + t) (BatchNorm+ReLU on the fly)
__device__ __forceinline__ void pack4_bnrelu(v16h& dst, int off, float4 v, float4 s, float4 t){
  dst[off+0] = (_Float16)fmaxf(fmaf(v.x, s.x, t.x), 0.f);
  dst[off+1] = (_Float16)fmaxf(fmaf(v.y, s.y, t.y), 0.f);
  dst[off+2] = (_Float16)fmaxf(fmaf(v.z, s.z, t.z), 0.f);
  dst[off+3] = (_Float16)fmaxf(fmaf(v.w, s.w, t.w), 0.f);
}

// ---------------- fill ----------------
__global__ void k_fill(float* __restrict__ p, int n, float v){
  int i = blockIdx.x * blockDim.x + threadIdx.x;
  if (i < n) p[i] = v;
}

// ---------------- node encoder: h = x @ W(64x64) + b  (WMMA f16) ----------------
// LDS holds W transposed (Wt[n][k]) so each B-fragment is two contiguous v8h loads (k..k+7, k+16..k+23).
__global__ void k_node_encode(const float* __restrict__ x, const float* __restrict__ W,
                              const float* __restrict__ bias, float* __restrict__ h)
{
  alignas(32) __shared__ _Float16 Wt[D * D];   // Wt[n*64 + k]
  __shared__ float bl[D];
  for (int i = threadIdx.x; i < D * D; i += blockDim.x){
    const int k = i >> 6, n = i & 63;          // W row-major [k][n]
    Wt[n * D + k] = (_Float16)W[i];
  }
  if (threadIdx.x < D) bl[threadIdx.x] = bias[threadIdx.x];
  __syncthreads();
  const int wave = threadIdx.x >> 5, lane = threadIdx.x & 31;
  const int tile = blockIdx.x * 8 + wave;
  if (tile >= N_NODES / 16) return;            // wave-uniform exit (EXEC stays full)
  const int row0 = tile * 16;
  const int mrow = lane & 15;
  const int kbA  = (lane < 16) ? 0 : 8;        // ISA A-frag K split
  const int kbB  = (lane < 16) ? 0 : 8;        // B-frag K split (same as A)
  const int n    = lane & 15;
  const float4* xr4 = (const float4*)(x + (size_t)(row0 + mrow) * D);
  const int b4 = kbA >> 2;
  float4 p0 = xr4[b4+0],  p1 = xr4[b4+1];      // K = kbA..kbA+7
  float4 p2 = xr4[b4+4],  p3 = xr4[b4+5];      // K = kbA+16..kbA+23
  float4 p4 = xr4[b4+8],  p5 = xr4[b4+9];      // +32
  float4 p6 = xr4[b4+12], p7 = xr4[b4+13];
  v16h a0, a1;
  pack8(a0, 0, p0, p1); pack8(a0, 8, p2, p3);
  pack8(a1, 0, p4, p5); pack8(a1, 8, p6, p7);
#pragma unroll
  for (int nt = 0; nt < 4; ++nt){
    const int col0 = nt * 16;
    const v16h b0 = ldB(&Wt[(col0 + n) * D + kbB]);
    const v16h b1 = ldB(&Wt[(col0 + n) * D + 32 + kbB]);
    v8f c = {};
    c = __builtin_amdgcn_wmma_f32_16x16x32_f16(false, a0, false, b0, (short)0, c, false, false);
    c = __builtin_amdgcn_wmma_f32_16x16x32_f16(false, a1, false, b1, (short)0, c, false, false);
    const float bc = bl[col0 + n];
#pragma unroll
    for (int v = 0; v < 8; ++v)
      h[(size_t)(row0 + v + ((lane >= 16) ? 8 : 0)) * D + col0 + n] = c[v] + bc;
  }
}

// ---------------- edge passes: recompute e on the fly; 3-pass segment softmax ----------------
// PASS 0: segment max (positive floats -> uint atomicMax); PASS 1: sum of exp; PASS 2: weighted agg
template<int PASS>
__global__ void k_edge(const int* __restrict__ ei, const float* __restrict__ eattr,
                       const float* __restrict__ We, const float* __restrict__ be,
                       const float* __restrict__ h, float* __restrict__ mbuf,
                       float* __restrict__ sbuf, float* __restrict__ agg)
{
  alignas(16) __shared__ float Wt[D * 16];     // Wt[c*16 + k], transposed edge weight
  __shared__ float bl[D];
  for (int i = threadIdx.x; i < D * 16; i += blockDim.x){
    const int k = i >> 6, c = i & 63;          // We row-major [k][c]
    Wt[c * 16 + k] = We[i];
  }
  if (threadIdx.x < D) bl[threadIdx.x] = be[threadIdx.x];
  __syncthreads();
  const int gid = blockIdx.x * blockDim.x + threadIdx.x;
  const int e   = gid >> 5;                    // one wave per edge
  if (e >= N_EDGES) return;
  const int lane = threadIdx.x & 31;
  const int src  = ei[e];
  const int dst  = ei[N_EDGES + e];
  if (lane == 0){
    int pe = (e + 64 < N_EDGES) ? (e + 64) : (N_EDGES - 1);
    __builtin_prefetch(eattr + (size_t)pe * 16, 0, 0);   // global_prefetch_b8
  }
  const float4* ea4 = (const float4*)(eattr + (size_t)e * 16);
  const float4 q0 = ea4[0], q1 = ea4[1], q2 = ea4[2], q3 = ea4[3];
  const int c0 = lane, c1 = lane + 32;
  const float4* w0 = (const float4*)&Wt[c0 * 16];
  const float4* w1 = (const float4*)&Wt[c1 * 16];
  float acc0 = bl[c0], acc1 = bl[c1];
  acc0 = dot4(q0, w0[0], acc0); acc0 = dot4(q1, w0[1], acc0);
  acc0 = dot4(q2, w0[2], acc0); acc0 = dot4(q3, w0[3], acc0);
  acc1 = dot4(q0, w1[0], acc1); acc1 = dot4(q1, w1[1], acc1);
  acc1 = dot4(q2, w1[2], acc1); acc1 = dot4(q3, w1[3], acc1);
  const float* hs = h + (size_t)src * D;       // gather, L2-resident (12.8 MB)
  const float m0 = fmaxf(hs[c0] + acc0, 0.f) + 1e-7f;
  const float m1 = fmaxf(hs[c1] + acc1, 0.f) + 1e-7f;
  const size_t o0 = (size_t)dst * D + c0, o1 = (size_t)dst * D + c1;
  if (PASS == 0){
    atomicMax((unsigned int*)&mbuf[o0], __float_as_uint(m0));
    atomicMax((unsigned int*)&mbuf[o1], __float_as_uint(m1));
  } else if (PASS == 1){
    atomicAdd(&sbuf[o0], __expf(m0 - mbuf[o0]));
    atomicAdd(&sbuf[o1], __expf(m1 - mbuf[o1]));
  } else {
    atomicAdd(&agg[o0], m0 * __expf(m0 - mbuf[o0]) / (sbuf[o0] + 1e-16f));
    atomicAdd(&agg[o1], m1 * __expf(m1 - mbuf[o1]) / (sbuf[o1] + 1e-16f));
  }
}

// ---------------- MessageNorm + residual (wave per node) ----------------
__global__ void k_msgnorm(const float* __restrict__ h, const float* __restrict__ agg,
                          const float* __restrict__ msg_scale, int layer,
                          float* __restrict__ outb)
{
  const int wave = threadIdx.x >> 5, lane = threadIdx.x & 31;
  const int node = blockIdx.x * 8 + wave;
  if (node >= N_NODES) return;
  const int c0 = lane, c1 = lane + 32;
  const float h0 = h[(size_t)node * D + c0], h1 = h[(size_t)node * D + c1];
  const float a0 = agg[(size_t)node * D + c0], a1 = agg[(size_t)node * D + c1];
  const float hn = sqrtf(wave_sum(h0 * h0 + h1 * h1));
  const float an = sqrtf(wave_sum(a0 * a0 + a1 * a1));
  const float k  = msg_scale[layer] * hn / fmaxf(an, 1e-12f);
  outb[(size_t)node * D + c0] = h0 + k * a0;
  outb[(size_t)node * D + c1] = h1 + k * a1;
}

// ---------------- MLP1: t1 = out @ W1(64x128) + b1  (WMMA) ----------------
__global__ void k_mlp1(const float* __restrict__ A, const float* __restrict__ W,
                       const float* __restrict__ bias, float* __restrict__ t1)
{
  alignas(32) __shared__ _Float16 Wt[HID * D];  // Wt[n*64 + k]
  __shared__ float bl[HID];
  for (int i = threadIdx.x; i < D * HID; i += blockDim.x){
    const int k = i >> 7, n = i & 127;          // W row-major [k][n], 64x128
    Wt[n * D + k] = (_Float16)W[i];
  }
  for (int i = threadIdx.x; i < HID; i += blockDim.x) bl[i] = bias[i];
  __syncthreads();
  const int wave = threadIdx.x >> 5, lane = threadIdx.x & 31;
  const int tile = blockIdx.x * 8 + wave;
  if (tile >= N_NODES / 16) return;
  const int row0 = tile * 16;
  const int mrow = lane & 15, kbA = (lane < 16) ? 0 : 8, kbB = (lane < 16) ? 0 : 8, n = lane & 15;
  const float4* ar4 = (const float4*)(A + (size_t)(row0 + mrow) * D);
  const int b4 = kbA >> 2;
  float4 p0 = ar4[b4+0],  p1 = ar4[b4+1];
  float4 p2 = ar4[b4+4],  p3 = ar4[b4+5];
  float4 p4 = ar4[b4+8],  p5 = ar4[b4+9];
  float4 p6 = ar4[b4+12], p7 = ar4[b4+13];
  v16h a0, a1;
  pack8(a0, 0, p0, p1); pack8(a0, 8, p2, p3);
  pack8(a1, 0, p4, p5); pack8(a1, 8, p6, p7);
#pragma unroll
  for (int nt = 0; nt < 8; ++nt){
    const int col0 = nt * 16;
    const v16h b0 = ldB(&Wt[(col0 + n) * D + kbB]);
    const v16h b1 = ldB(&Wt[(col0 + n) * D + 32 + kbB]);
    v8f c = {};
    c = __builtin_amdgcn_wmma_f32_16x16x32_f16(false, a0, false, b0, (short)0, c, false, false);
    c = __builtin_amdgcn_wmma_f32_16x16x32_f16(false, a1, false, b1, (short)0, c, false, false);
    const float bc = bl[col0 + n];
#pragma unroll
    for (int v = 0; v < 8; ++v)
      t1[(size_t)(row0 + v + ((lane >= 16) ? 8 : 0)) * HID + col0 + n] = c[v] + bc;
  }
}

// ---------------- BatchNorm stats over 50000 rows x 128 cols ----------------
__global__ void k_bn_stats(const float* __restrict__ t1, float* __restrict__ acc)
{
  const int col  = threadIdx.x & 127;
  const int half = threadIdx.x >> 7;
  const int r0   = blockIdx.x * 400;             // 125 blocks * 400 rows = 50000
  float s = 0.f, q = 0.f;
  for (int r = r0 + half; r < r0 + 400; r += 2){
    const float v = t1[(size_t)r * HID + col];
    s += v; q += v * v;
  }
  __shared__ float sh[256], qh[256];
  sh[threadIdx.x] = s; qh[threadIdx.x] = q;
  __syncthreads();
  if (half == 0){
    atomicAdd(&acc[col],       sh[col] + sh[col + 128]);
    atomicAdd(&acc[HID + col], qh[col] + qh[col + 128]);
  }
}

__global__ void k_bn_finalize(const float* __restrict__ acc, const float* __restrict__ g,
                              const float* __restrict__ b, float* __restrict__ scale,
                              float* __restrict__ shift, int C, float invN)
{
  const int c = blockIdx.x * blockDim.x + threadIdx.x;
  if (c >= C) return;
  const float mean = acc[c] * invN;
  const float var  = acc[C + c] * invN - mean * mean;
  const float rs   = rsqrtf(var + 1e-5f);
  const float sc   = g[c] * rs;
  scale[c] = sc;
  shift[c] = b[c] - mean * sc;
}

// ---------------- MLP2: t2 = relu(BN(t1)) @ W2(128x64) + b2  (BN fused into A-frag) ----------------
__global__ void k_mlp2(const float* __restrict__ T1, const float* __restrict__ bnsc,
                       const float* __restrict__ bnsh, const float* __restrict__ W,
                       const float* __restrict__ bias, float* __restrict__ t2)
{
  alignas(32) __shared__ _Float16 Wt[D * HID];   // Wt[n*128 + k]
  alignas(16) __shared__ float scl[HID], shl[HID], bl[D];
  for (int i = threadIdx.x; i < HID * D; i += blockDim.x){
    const int k = i >> 6, n = i & 63;            // W row-major [k][n], 128x64
    Wt[n * HID + k] = (_Float16)W[i];
  }
  for (int i = threadIdx.x; i < HID; i += blockDim.x){ scl[i] = bnsc[i]; shl[i] = bnsh[i]; }
  if (threadIdx.x < D) bl[threadIdx.x] = bias[threadIdx.x];
  __syncthreads();
  const int wave = threadIdx.x >> 5, lane = threadIdx.x & 31;
  const int tile = blockIdx.x * 8 + wave;
  if (tile >= N_NODES / 16) return;
  const int row0 = tile * 16;
  const int mrow = lane & 15, kbA = (lane < 16) ? 0 : 8, kbB = (lane < 16) ? 0 : 8, n = lane & 15;
  const float4* ar4 = (const float4*)(T1 + (size_t)(row0 + mrow) * HID);
  const float4* sc4 = (const float4*)scl;
  const float4* sh4 = (const float4*)shl;
  v16h a[4];
#pragma unroll
  for (int kc = 0; kc < 4; ++kc){
    const int i0 = (kc * 32 + kbA) >> 2;          // first run: K = kc*32+kbA .. +7
    const int i1 = (kc * 32 + kbA + 16) >> 2;     // second:    K = kc*32+kbA+16 .. +23
    pack4_bnrelu(a[kc], 0,  ar4[i0],   sc4[i0],   sh4[i0]);
    pack4_bnrelu(a[kc], 4,  ar4[i0+1], sc4[i0+1], sh4[i0+1]);
    pack4_bnrelu(a[kc], 8,  ar4[i1],   sc4[i1],   sh4[i1]);
    pack4_bnrelu(a[kc], 12, ar4[i1+1], sc4[i1+1], sh4[i1+1]);
  }
#pragma unroll
  for (int nt = 0; nt < 4; ++nt){
    const int col0 = nt * 16;
    v8f c = {};
#pragma unroll
    for (int kc = 0; kc < 4; ++kc){
      const v16h bf = ldB(&Wt[(col0 + n) * HID + kc * 32 + kbB]);
      c = __builtin_amdgcn_wmma_f32_16x16x32_f16(false, a[kc], false, bf, (short)0, c, false, false);
    }
    const float bc = bl[col0 + n];
#pragma unroll
    for (int v = 0; v < 8; ++v)
      t2[(size_t)(row0 + v + ((lane >= 16) ? 8 : 0)) * D + col0 + n] = c[v] + bc;
  }
}

// ---------------- LayerNorm -> relu -> softmax -> readout accumulate (wave per node) ----------------
__global__ void k_ln_softmax(const float* __restrict__ t2, const float* __restrict__ g,
                             const float* __restrict__ b, float* __restrict__ readout)
{
  const int wave = threadIdx.x >> 5, lane = threadIdx.x & 31;
  const int node = blockIdx.x * 8 + wave;
  if (node >= N_NODES) return;
  const int c0 = lane, c1 = lane + 32;
  const float v0 = t2[(size_t)node * D + c0], v1 = t2[(size_t)node * D + c1];
  const float mu  = wave_sum(v0 + v1) * (1.f / 64.f);
  const float d0 = v0 - mu, d1 = v1 - mu;
  const float var = wave_sum(d0 * d0 + d1 * d1) * (1.f / 64.f);
  const float rs  = rsqrtf(var + 1e-5f);
  const float y0 = fmaxf(d0 * rs * g[c0] + b[c0], 0.f);
  const float y1 = fmaxf(d1 * rs * g[c1] + b[c1], 0.f);
  const float mx = wave_max(fmaxf(y0, y1));
  const float e0 = __expf(y0 - mx), e1 = __expf(y1 - mx);
  const float inv = 1.f / wave_sum(e0 + e1);
  readout[(size_t)node * D + c0] += e0 * inv;
  readout[(size_t)node * D + c1] += e1 * inv;
}

// ---------------- global add pool ----------------
__global__ void k_pool(const float* __restrict__ readout, const int* __restrict__ batch,
                       float* __restrict__ pooled)
{
  const int i = blockIdx.x * blockDim.x + threadIdx.x;
  if (i >= N_NODES * D) return;
  const int node = i >> 6, c = i & 63;
  atomicAdd(&pooled[(size_t)batch[node] * D + c], readout[i]);
}

// ---------------- classifier helpers (256 rows, tiny) ----------------
template<int ACT>   // 0 = relu, 1 = sigmoid
__global__ void k_fc(const float* __restrict__ A, const float* __restrict__ W,
                     const float* __restrict__ bias, float* __restrict__ Cm,
                     int M, int K, int N)
{
  const int i = blockIdx.x * blockDim.x + threadIdx.x;
  if (i >= M * N) return;
  const int r = i / N, c = i % N;
  float acc = bias[c];
  for (int k = 0; k < K; ++k) acc = fmaf(A[(size_t)r * K + k], W[(size_t)k * N + c], acc);
  Cm[i] = (ACT == 0) ? fmaxf(acc, 0.f) : 1.f / (1.f + __expf(-acc));
}

__global__ void k_colstats(const float* __restrict__ A, const float* __restrict__ g,
                           const float* __restrict__ b, float* __restrict__ scale,
                           float* __restrict__ shift, int rows, int C)
{
  const int c = blockIdx.x * blockDim.x + threadIdx.x;
  if (c >= C) return;
  float s = 0.f, q = 0.f;
  for (int r = 0; r < rows; ++r){ const float v = A[(size_t)r * C + c]; s += v; q += v * v; }
  const float mean = s / rows, var = q / rows - mean * mean;
  const float rs = rsqrtf(var + 1e-5f);
  const float sc = g[c] * rs;
  scale[c] = sc;
  shift[c] = b[c] - mean * sc;
}

__global__ void k_affine(float* __restrict__ A, const float* __restrict__ scale,
                         const float* __restrict__ shift, int n, int C)
{
  const int i = blockIdx.x * blockDim.x + threadIdx.x;
  if (i >= n) return;
  const int c = i % C;
  A[i] = fmaf(A[i], scale[c], shift[c]);
}

// =======================================================================
extern "C" void kernel_launch(void* const* d_in, const int* in_sizes, int n_in,
                              void* d_out, int out_size, void* d_ws, size_t ws_size,
                              hipStream_t stream)
{
  (void)in_sizes; (void)n_in; (void)out_size; (void)ws_size;
  const float* x          = (const float*)d_in[0];
  const int*   edge_index = (const int*)  d_in[1];
  const float* edge_attr  = (const float*)d_in[2];
  const int*   batch      = (const int*)  d_in[3];
  const float* node_W = (const float*)d_in[4];
  const float* node_b = (const float*)d_in[5];
  const float* edge_W = (const float*)d_in[6];
  const float* edge_b = (const float*)d_in[7];
  const float* mlp_W1 = (const float*)d_in[8];
  const float* mlp_b1 = (const float*)d_in[9];
  const float* bn_g   = (const float*)d_in[10];
  const float* bn_b   = (const float*)d_in[11];
  const float* mlp_W2 = (const float*)d_in[12];
  const float* mlp_b2 = (const float*)d_in[13];
  const float* ln_g   = (const float*)d_in[14];
  const float* ln_b   = (const float*)d_in[15];
  const float* msg_scale = (const float*)d_in[16];
  const float* cW1 = (const float*)d_in[17]; const float* cb1  = (const float*)d_in[18];
  const float* cg1 = (const float*)d_in[19]; const float* cbe1 = (const float*)d_in[20];
  const float* cW2 = (const float*)d_in[21]; const float* cb2  = (const float*)d_in[22];
  const float* cg2 = (const float*)d_in[23]; const float* cbe2 = (const float*)d_in[24];
  const float* cW3 = (const float*)d_in[25]; const float* cb3  = (const float*)d_in[26];
  const float* cg3 = (const float*)d_in[27]; const float* cbe3 = (const float*)d_in[28];
  const float* cW4 = (const float*)d_in[29]; const float* cb4  = (const float*)d_in[30];

  char* wptr = (char*)d_ws;
  auto carve = [&](size_t nf) -> float* {
    float* p = (float*)wptr;
    wptr += ((nf * sizeof(float) + 255) & ~(size_t)255);
    return p;
  };
  float* h       = carve((size_t)N_NODES * D);
  float* outb    = carve((size_t)N_NODES * D);
  float* mbuf    = carve((size_t)N_NODES * D);
  float* sbuf    = carve((size_t)N_NODES * D);
  float* aggb    = carve((size_t)N_NODES * D);
  float* t1      = carve((size_t)N_NODES * HID);
  float* t2      = carve((size_t)N_NODES * D);
  float* readout = carve((size_t)N_NODES * D);
  float* bn_acc  = carve(2 * HID);
  float* bn_sc   = carve(HID);
  float* bn_sh   = carve(HID);
  float* pooled  = carve((size_t)N_GRAPHS * D);
  float* g1      = carve((size_t)N_GRAPHS * 256);
  float* g2      = carve((size_t)N_GRAPHS * 128);
  float* g3      = carve((size_t)N_GRAPHS * 64);
  float* cs_sc   = carve(256);
  float* cs_sh   = carve(256);

  const int ND = N_NODES * D;
  k_fill<<<(ND + 255) / 256, 256, 0, stream>>>(readout, ND, 0.f);

  const int gemm_blocks = (N_NODES / 16 + 7) / 8;   // 391: 8 waves/block, 16 rows/wave
  const int node_blocks = N_NODES / 8;              // 6250: wave per node
  const int edge_blocks = N_EDGES / 8;              // 100000: wave per edge

  for (int l = 0; l < 3; ++l){
    k_node_encode<<<gemm_blocks, 256, 0, stream>>>(x, node_W + (size_t)l * D * D, node_b + l * D, h);
    k_fill<<<(ND + 255) / 256, 256, 0, stream>>>(mbuf, ND, 0.f);
    k_fill<<<(ND + 255) / 256, 256, 0, stream>>>(sbuf, ND, 0.f);
    k_fill<<<(ND + 255) / 256, 256, 0, stream>>>(aggb, ND, 0.f);
    const float* We = edge_W + (size_t)l * 16 * D;
    const float* be = edge_b + l * D;
    k_edge<0><<<edge_blocks, 256, 0, stream>>>(edge_index, edge_attr, We, be, h, mbuf, sbuf, aggb);
    k_edge<1><<<edge_blocks, 256, 0, stream>>>(edge_index, edge_attr, We, be, h, mbuf, sbuf, aggb);
    k_edge<2><<<edge_blocks, 256, 0, stream>>>(edge_index, edge_attr, We, be, h, mbuf, sbuf, aggb);
    k_msgnorm<<<node_blocks, 256, 0, stream>>>(h, aggb, msg_scale, l, outb);
    k_mlp1<<<gemm_blocks, 256, 0, stream>>>(outb, mlp_W1 + (size_t)l * D * HID, mlp_b1 + l * HID, t1);
    k_fill<<<1, 256, 0, stream>>>(bn_acc, 2 * HID, 0.f);
    k_bn_stats<<<125, 256, 0, stream>>>(t1, bn_acc);
    k_bn_finalize<<<1, 128, 0, stream>>>(bn_acc, bn_g + l * HID, bn_b + l * HID, bn_sc, bn_sh, HID,
                                         1.f / (float)N_NODES);
    k_mlp2<<<gemm_blocks, 256, 0, stream>>>(t1, bn_sc, bn_sh, mlp_W2 + (size_t)l * HID * D,
                                            mlp_b2 + l * D, t2);
    k_ln_softmax<<<node_blocks, 256, 0, stream>>>(t2, ln_g + l * D, ln_b + l * D, readout);
  }

  k_fill<<<(N_GRAPHS * D + 255) / 256, 256, 0, stream>>>(pooled, N_GRAPHS * D, 0.f);
  k_pool<<<(ND + 255) / 256, 256, 0, stream>>>(readout, batch, pooled);

  k_fc<0><<<(N_GRAPHS * 256 + 255) / 256, 256, 0, stream>>>(pooled, cW1, cb1, g1, N_GRAPHS, 64, 256);
  k_colstats<<<1, 256, 0, stream>>>(g1, cg1, cbe1, cs_sc, cs_sh, N_GRAPHS, 256);
  k_affine<<<(N_GRAPHS * 256 + 255) / 256, 256, 0, stream>>>(g1, cs_sc, cs_sh, N_GRAPHS * 256, 256);

  k_fc<0><<<(N_GRAPHS * 128 + 255) / 256, 256, 0, stream>>>(g1, cW2, cb2, g2, N_GRAPHS, 256, 128);
  k_colstats<<<1, 128, 0, stream>>>(g2, cg2, cbe2, cs_sc, cs_sh, N_GRAPHS, 128);
  k_affine<<<(N_GRAPHS * 128 + 255) / 256, 256, 0, stream>>>(g2, cs_sc, cs_sh, N_GRAPHS * 128, 128);

  k_fc<0><<<(N_GRAPHS * 64 + 255) / 256, 256, 0, stream>>>(g2, cW3, cb3, g3, N_GRAPHS, 128, 64);
  k_colstats<<<1, 64, 0, stream>>>(g3, cg3, cbe3, cs_sc, cs_sh, N_GRAPHS, 64);
  k_affine<<<(N_GRAPHS * 64 + 255) / 256, 256, 0, stream>>>(g3, cs_sc, cs_sh, N_GRAPHS * 64, 64);

  k_fc<1><<<(N_GRAPHS * OUT_DIM + 255) / 256, 256, 0, stream>>>(g3, cW4, cb4, (float*)d_out,
                                                                N_GRAPHS, 64, OUT_DIM);
}
